// Block_309237645595
// MI455X (gfx1250) — matrix operand order repaired and checked by reading, not compiled, not run on hardware
//
#include <hip/hip_runtime.h>
#include <hip/hip_bf16.h>
#include <cstdint>

// Problem constants (from reference)
#define T_TOK  8192      // B*S
#define BATCH  4
#define SEQ    2048
#define DMODEL 1024
#define NHEAD  8
#define DHEAD  128
#define DFF_   4096
#define EPSF   1e-6f

typedef __attribute__((ext_vector_type(16))) __bf16       v16bf;
typedef __attribute__((ext_vector_type(8)))  float        v8f;
typedef __attribute__((ext_vector_type(8)))  unsigned int v8u;

#if __has_builtin(__builtin_amdgcn_tensor_load_to_lds) && __has_builtin(__builtin_amdgcn_s_wait_tensorcnt)
#define HAVE_TDM 1
typedef __attribute__((ext_vector_type(4))) unsigned int u32x4;
typedef __attribute__((ext_vector_type(8))) int          i32x8;
typedef __attribute__((ext_vector_type(4))) int          i32x4;
#endif

__device__ __forceinline__ unsigned short f2bf(float f) {
  unsigned u = __builtin_bit_cast(unsigned, f);
  u += 0x7FFFu + ((u >> 16) & 1u);          // round-to-nearest-even
  return (unsigned short)(u >> 16);
}
__device__ __forceinline__ float bf2f(unsigned short h) {
  return __builtin_bit_cast(float, ((unsigned)h) << 16);
}
__device__ __forceinline__ float sigm(float x) { return 1.0f / (1.0f + __expf(-x)); }

// ---------------------------------------------------------------------------
// f32 -> bf16 (plain)
// ---------------------------------------------------------------------------
__global__ void cvt_f32_bf16(const float* __restrict__ src,
                             unsigned short* __restrict__ dst, int n) {
  int i = blockIdx.x * blockDim.x + threadIdx.x;
  if (i < n) dst[i] = f2bf(src[i]);
}

// ---------------------------------------------------------------------------
// f32 [K][N] -> bf16 transposed [N][K]. 32x32 tiles, block=256 (32x8).
// Weights pre-transposed so GEMM B-tiles are contiguous-in-K like A-tiles.
// ---------------------------------------------------------------------------
__global__ __launch_bounds__(256) void cvt_transpose_bf16(
    const float* __restrict__ src, unsigned short* __restrict__ dst,
    int K, int N) {
  __shared__ float tile[32][33];
  int tx = threadIdx.x & 31, ty = threadIdx.x >> 5;
  int n0 = blockIdx.x * 32, k0 = blockIdx.y * 32;
  #pragma unroll
  for (int i = 0; i < 32; i += 8)
    tile[ty + i][tx] = src[(size_t)(k0 + ty + i) * N + n0 + tx];
  __syncthreads();
  #pragma unroll
  for (int i = 0; i < 32; i += 8)
    dst[(size_t)(n0 + ty + i) * K + k0 + tx] = f2bf(tile[tx][ty + i]);
}

// ---------------------------------------------------------------------------
// Per-token RMSNorm (D=1024) -> bf16 output. block=256, grid=T
// ---------------------------------------------------------------------------
__global__ __launch_bounds__(256) void rmsnorm_to_bf16(
    const float* __restrict__ x, unsigned short* __restrict__ out) {
  int t = blockIdx.x, tid = threadIdx.x;
  const float4* xr = (const float4*)(x + (size_t)t * DMODEL);
  float4 v4 = xr[tid];
  float ss = v4.x * v4.x + v4.y * v4.y + v4.z * v4.z + v4.w * v4.w;
  #pragma unroll
  for (int off = 16; off; off >>= 1) ss += __shfl_xor(ss, off, 32);
  __shared__ float part[8];
  if ((tid & 31) == 0) part[tid >> 5] = ss;
  __syncthreads();
  float tot = 0.f;
  #pragma unroll
  for (int w = 0; w < 8; ++w) tot += part[w];
  float inv = rsqrtf(tot / (float)DMODEL + EPSF);
  unsigned short* o = out + (size_t)t * DMODEL + tid * 4;
  o[0] = f2bf(v4.x * inv); o[1] = f2bf(v4.y * inv);
  o[2] = f2bf(v4.z * inv); o[3] = f2bf(v4.w * inv);
}

// ---------------------------------------------------------------------------
// WMMA bf16 GEMM: C[M,N] = A[M,K] * Bt[N,K]^T   (Bt is pre-transposed weights)
//   256 threads / 8 waves; block tile 128x128, K-step 32
//   waves 4(M) x 2(N); wave tile 32x64 = 2x4 wmma tiles
//   Tile staging via Tensor Data Mover (tensor_load_to_lds):
//   2D descriptor, tile 32(k) x 128(rows), pad 4 DWORDs per 16 DWORDs
//   -> LDS row stride 40 bf16 (16B-aligned rows for ds_load_b128).
// ---------------------------------------------------------------------------
#ifdef HAVE_TDM
__device__ __forceinline__ void tdm_load_tile_128x32(
    unsigned lds_off, const unsigned short* gsrc,
    unsigned rowStrideElems, unsigned rowsTensor, unsigned kTensor) {
  unsigned long long ga = (unsigned long long)(size_t)gsrc;
  u32x4 g0;
  g0[0] = 1u;                                            // count=1 (valid D#)
  g0[1] = lds_off;                                       // lds_addr (bytes)
  g0[2] = (unsigned)(ga & 0xFFFFFFFFu);                  // global_addr[31:0]
  g0[3] = (unsigned)((ga >> 32) & 0x01FFFFFFu) | (2u << 30);  // addr[56:32] | type=2
  i32x8 g1;
  g1[0] = (1 << 16) | (1 << 20) | (3 << 22) | (3 << 25); // data_size=2B, pad_en, pad_interval=16DW, pad_amount=4DW
  g1[1] = (int)((kTensor & 0xFFFFu) << 16);              // tensor_dim0[15:0]
  g1[2] = (int)(((kTensor >> 16) & 0xFFFFu) | ((rowsTensor & 0xFFFFu) << 16)); // dim0 hi | dim1 lo
  g1[3] = (int)(((rowsTensor >> 16) & 0xFFFFu) | (32u << 16));                 // dim1 hi | tile_dim0=32
  g1[4] = (int)128u;                                     // tile_dim1=128, tile_dim2=0
  g1[5] = (int)rowStrideElems;                           // tensor_dim0_stride lo32
  g1[6] = 0;                                             // stride hi16 | dim1_stride lo16
  g1[7] = 0;
  i32x4 z4 = {0, 0, 0, 0};
  i32x8 z8 = {0, 0, 0, 0, 0, 0, 0, 0};
  __builtin_amdgcn_tensor_load_to_lds(g0, g1, z4, z4, z8, 0);
}
#endif

__device__ __forceinline__ void stage_tile_sync(
    const unsigned short* __restrict__ g, unsigned short* lds, int tid,
    size_t rowStrideElems, int r0, int k0) {
  #pragma unroll
  for (int p = 0; p < 2; ++p) {
    int c = tid + p * 256, row = c >> 2, kp = (c & 3) << 3;
    uint4 v = *(const uint4*)(g + (size_t)(r0 + row) * rowStrideElems + k0 + kp);
    *(uint4*)&lds[row * 40 + kp] = v;
  }
}

__global__ __launch_bounds__(256) void gemm_bf16_wmma(
    const unsigned short* __restrict__ A,   // [M,K] bf16
    const unsigned short* __restrict__ Bt,  // [N,K] bf16 (transposed weights)
    float* __restrict__ outF,
    unsigned short* __restrict__ outB,
    const float* __restrict__ resid,
    int M, int N, int K, int relu2) {
  __shared__ __align__(16) unsigned short lds_a[128 * 40];  // [m][k], stride 40
  __shared__ __align__(16) unsigned short lds_b[128 * 40];  // [n][k], stride 40

  const int tid  = threadIdx.x;
  const int lane = tid & 31;
  const int wave = tid >> 5;
  const int wm   = wave & 3;   // 4 waves along M
  const int wn   = wave >> 2;  // 2 waves along N
  const int half = lane >> 4;
  const int l15  = lane & 15;

  const int n0 = blockIdx.x * 128;
  const int m0 = blockIdx.y * 128;

#ifdef HAVE_TDM
  const unsigned ldsA_off = (unsigned)(size_t)&lds_a[0];  // flat addr low 32 = LDS offset
  const unsigned ldsB_off = (unsigned)(size_t)&lds_b[0];
#endif

  v8f acc[2][4];
  #pragma unroll
  for (int mi = 0; mi < 2; ++mi)
    #pragma unroll
    for (int ni = 0; ni < 4; ++ni)
      #pragma unroll
      for (int r = 0; r < 8; ++r) acc[mi][ni][r] = 0.0f;

  for (int k0 = 0; k0 < K; k0 += 32) {
#ifdef HAVE_TDM
    if (tid < 32) {   // wave 0 drives the Tensor Data Mover
      tdm_load_tile_128x32(ldsA_off, A  + (size_t)m0 * K + k0, (unsigned)K,
                           (unsigned)M, (unsigned)K);
      tdm_load_tile_128x32(ldsB_off, Bt + (size_t)n0 * K + k0, (unsigned)K,
                           (unsigned)N, (unsigned)K);
      __builtin_amdgcn_s_wait_tensorcnt(0);
    }
    __syncthreads();
#else
    stage_tile_sync(A,  lds_a, tid, (size_t)K, m0, k0);
    stage_tile_sync(Bt, lds_b, tid, (size_t)K, n0, k0);
    if (k0 + 32 < K) {
      int row = tid >> 2, kp = (tid & 3) << 3;
      __builtin_prefetch(A  + (size_t)(m0 + row) * K + k0 + 32 + kp, 0, 0);
      __builtin_prefetch(Bt + (size_t)(n0 + row) * K + k0 + 32 + kp, 0, 0);
    }
    __syncthreads();
#endif

    // ---- build fragments (ISA 16-bit A/B layouts, wave32)
    v16bf afr[2], bfr[4];
    #pragma unroll
    for (int mi = 0; mi < 2; ++mi) {
      int arow = wm * 32 + mi * 16 + l15;
      v8u au;
      #pragma unroll
      for (int j = 0; j < 8; ++j) {
        int kk = ((j >> 2) << 4) + (half << 3) + ((j & 3) << 1);
        au[j] = *(const unsigned*)&lds_a[arow * 40 + kk];
      }
      afr[mi] = __builtin_bit_cast(v16bf, au);
    }
    #pragma unroll
    for (int ni = 0; ni < 4; ++ni) {
      int bcol = wn * 64 + ni * 16 + l15;
      v8u bu;
      #pragma unroll
      for (int j = 0; j < 8; ++j) {
        int kk = ((j >> 2) << 4) + (half << 3) + ((j & 3) << 1);
        bu[j] = *(const unsigned*)&lds_b[bcol * 40 + kk];
      }
      bfr[ni] = __builtin_bit_cast(v16bf, bu);
    }
    #pragma unroll
    for (int mi = 0; mi < 2; ++mi)
      #pragma unroll
      for (int ni = 0; ni < 4; ++ni)
        acc[mi][ni] = __builtin_amdgcn_wmma_f32_16x16x32_bf16(
            false, afr[mi], false, bfr[ni], (short)0, acc[mi][ni], false, false);
    __syncthreads();
  }

  // ---- epilogue (C layout: VGPR r -> M=r + 8*half, N=l15)
  #pragma unroll
  for (int mi = 0; mi < 2; ++mi)
    #pragma unroll
    for (int ni = 0; ni < 4; ++ni)
      #pragma unroll
      for (int r = 0; r < 8; ++r) {
        int row = m0 + wm * 32 + mi * 16 + r + (half << 3);
        int col = n0 + wn * 64 + ni * 16 + l15;
        size_t idx = (size_t)row * N + col;
        float v = acc[mi][ni][r];
        if (relu2) {
          float t = v > 0.0f ? v : 0.0f;
          outB[idx] = f2bf(t * t);
        } else {
          if (resid) v += resid[idx];
          outF[idx] = v;
        }
      }
}

// ---------------------------------------------------------------------------
// beta / alpha projections (N=8 each) + nonlinearity. grid=T, block=256
// ---------------------------------------------------------------------------
__global__ __launch_bounds__(256) void proj_beta_alpha(
    const unsigned short* __restrict__ m,  // [T,D] bf16 (rmsnormed x)
    const float* __restrict__ Wb, const float* __restrict__ Wa,
    const float* __restrict__ A_log, const float* __restrict__ dt_bias,
    float* __restrict__ beta, float* __restrict__ alpha) {
  int t = blockIdx.x, tid = threadIdx.x;
  int h = tid >> 5, lane = tid & 31;
  float accB = 0.f, accA = 0.f;
  #pragma unroll 4
  for (int j = 0; j < 32; ++j) {
    int idx = j * 32 + lane;
    float xm = bf2f(m[(size_t)t * DMODEL + idx]);
    accB += xm * Wb[idx * NHEAD + h];
    accA += xm * Wa[idx * NHEAD + h];
  }
  #pragma unroll
  for (int off = 16; off; off >>= 1) {
    accB += __shfl_xor(accB, off, 32);
    accA += __shfl_xor(accA, off, 32);
  }
  if (lane == 0) {
    beta[t * NHEAD + h] = sigm(accB);
    float z  = accA + dt_bias[h];
    float sp = (z > 20.0f) ? z : log1pf(__expf(z));
    alpha[t * NHEAD + h] = __expf(-__expf(A_log[h]) * sp);
  }
}

// ---------------------------------------------------------------------------
// silu (+ optional l2norm over 128). grid = T*H, block = 128. In-place.
// ---------------------------------------------------------------------------
__global__ __launch_bounds__(128) void act_silu_l2(float* __restrict__ p, int do_norm) {
  int rh = blockIdx.x, tid = threadIdx.x;
  size_t idx = (size_t)rh * DHEAD + tid;
  float x = p[idx];
  float s = x * sigm(x);
  if (do_norm) {
    __shared__ float red[128];
    red[tid] = s * s;
    __syncthreads();
    for (int off = 64; off; off >>= 1) {
      if (tid < off) red[tid] += red[tid + off];
      __syncthreads();
    }
    s *= rsqrtf(red[0] + EPSF);   // l2norm: sum (not mean) + eps
  }
  p[idx] = s;
}

// ---------------------------------------------------------------------------
// Gated delta-rule scan. grid = B*H (32), block = 256.
// State S[128 dk][128 dv] f32 in registers: thread (e = tid>>1, dh = tid&1)
// owns S[dh*64 .. dh*64+63][e]; pair combined with shfl_xor(1).
// ---------------------------------------------------------------------------
__global__ __launch_bounds__(256) void delta_scan(
    const float* __restrict__ q, const float* __restrict__ k,
    const float* __restrict__ v, const float* __restrict__ alpha,
    const float* __restrict__ beta, float* __restrict__ o) {
  const int bh = blockIdx.x;
  const int b = bh >> 3, h = bh & 7;
  const int tid = threadIdx.x;
  const int e = tid >> 1, dh = tid & 1;
  const float scale = 0.08838834764831845f;  // 1/sqrt(128)

  __shared__ float lq[128], lk[128], lv[128], sab[2];
  float S[64];
  #pragma unroll
  for (int i = 0; i < 64; ++i) S[i] = 0.f;

  for (int s = 0; s < SEQ; ++s) {
    size_t tok  = (size_t)b * SEQ + s;
    size_t base = tok * (NHEAD * DHEAD) + (size_t)h * DHEAD;
    if (tid < 128) { lq[tid] = q[base + tid]; lv[tid] = v[base + tid]; }
    else           { lk[tid - 128] = k[base + tid - 128]; }
    if (tid == 0) { sab[0] = alpha[tok * NHEAD + h]; sab[1] = beta[tok * NHEAD + h]; }
    __syncthreads();

    float a = sab[0], bt = sab[1];
    const float* kk = &lk[dh * 64];
    const float* qq = &lq[dh * 64];

    float dk_ = 0.f;
    #pragma unroll
    for (int i = 0; i < 64; ++i) dk_ += kk[i] * S[i];
    dk_ *= a;                                   // k . (a*S_old)
    float kv    = dk_ + __shfl_xor(dk_, 1, 32);
    float delta = (lv[e] - kv) * bt;

    float dq = 0.f;
    #pragma unroll
    for (int i = 0; i < 64; ++i) {
      float sn = a * S[i] + kk[i] * delta;      // S_new
      S[i] = sn;
      dq += qq[i] * sn;
    }
    float ov = dq + __shfl_xor(dq, 1, 32);
    if (dh == 0) o[base + e] = ov * scale;
    __syncthreads();
  }
}

// ---------------------------------------------------------------------------
// Gated RMSNorm: out = rms_norm(o)*w*silu(gate) -> bf16. grid=T*H, block=128
// ---------------------------------------------------------------------------
__global__ __launch_bounds__(128) void gated_rmsnorm(
    const float* __restrict__ o, const float* __restrict__ gpre,
    const float* __restrict__ w, unsigned short* __restrict__ out) {
  int rh = blockIdx.x, tid = threadIdx.x;
  size_t idx = (size_t)rh * DHEAD + tid;
  float ov = o[idx];
  __shared__ float red[128];
  red[tid] = ov * ov;
  __syncthreads();
  for (int off = 64; off; off >>= 1) {
    if (tid < off) red[tid] += red[tid + off];
    __syncthreads();
  }
  float inv = rsqrtf(red[0] / (float)DHEAD + EPSF);
  float g = gpre[idx];
  out[idx] = f2bf(ov * inv * w[tid] * (g * sigm(g)));
}

// ---------------------------------------------------------------------------
// Orchestration
// ---------------------------------------------------------------------------
extern "C" void kernel_launch(void* const* d_in, const int* in_sizes, int n_in,
                              void* d_out, int out_size, void* d_ws, size_t ws_size,
                              hipStream_t stream) {
  (void)in_sizes; (void)n_in; (void)out_size; (void)ws_size;
  const float* x      = (const float*)d_in[0];
  const float* Wq     = (const float*)d_in[1];
  const float* Wk     = (const float*)d_in[2];
  const float* Wv     = (const float*)d_in[3];
  const float* Wb     = (const float*)d_in[4];
  const float* Wa     = (const float*)d_in[5];
  const float* A_log  = (const float*)d_in[6];
  const float* dtb    = (const float*)d_in[7];
  const float* Wg     = (const float*)d_in[8];
  const float* onw    = (const float*)d_in[9];
  const float* Wo     = (const float*)d_in[10];
  const float* Wfc    = (const float*)d_in[11];
  const float* Wpr    = (const float*)d_in[12];
  float* out = (float*)d_out;

  char* ws = (char*)d_ws;
  size_t off = 0;
  auto alloc = [&](size_t bytes) -> char* {
    char* p = ws + off;
    off = (off + bytes + 255) & ~(size_t)255;
    return p;
  };
  const size_t TD = (size_t)T_TOK * DMODEL;
  // transposed bf16 weights: [N][K]
  unsigned short* wq16  = (unsigned short*)alloc((size_t)DMODEL * DMODEL * 2);
  unsigned short* wk16  = (unsigned short*)alloc((size_t)DMODEL * DMODEL * 2);
  unsigned short* wv16  = (unsigned short*)alloc((size_t)DMODEL * DMODEL * 2);
  unsigned short* wg16  = (unsigned short*)alloc((size_t)DMODEL * DMODEL * 2);
  unsigned short* wo16  = (unsigned short*)alloc((size_t)DMODEL * DMODEL * 2);
  unsigned short* wfc16 = (unsigned short*)alloc((size_t)DMODEL * DFF_ * 2);
  unsigned short* wpr16 = (unsigned short*)alloc((size_t)DFF_ * DMODEL * 2);
  unsigned short* m0    = (unsigned short*)alloc(TD * 2);
  float* qb   = (float*)alloc(TD * 4);
  float* kb   = (float*)alloc(TD * 4);
  float* vb   = (float*)alloc(TD * 4);
  float* gb   = (float*)alloc(TD * 4);
  float* beta = (float*)alloc((size_t)T_TOK * NHEAD * 4);
  float* alph = (float*)alloc((size_t)T_TOK * NHEAD * 4);
  float* ob   = (float*)alloc(TD * 4);
  unsigned short* og16 = (unsigned short*)alloc(TD * 2);
  float* x1   = (float*)alloc(TD * 4);
  unsigned short* m1   = (unsigned short*)alloc(TD * 2);
  unsigned short* ff16 = (unsigned short*)alloc((size_t)T_TOK * DFF_ * 2);

  // 1) weights -> bf16, transposed to [N][K]
  {
    dim3 blk(256);
    dim3 gDD(DMODEL / 32, DMODEL / 32);       // K=DMODEL, N=DMODEL
    cvt_transpose_bf16<<<gDD, blk, 0, stream>>>(Wq, wq16, DMODEL, DMODEL);
    cvt_transpose_bf16<<<gDD, blk, 0, stream>>>(Wk, wk16, DMODEL, DMODEL);
    cvt_transpose_bf16<<<gDD, blk, 0, stream>>>(Wv, wv16, DMODEL, DMODEL);
    cvt_transpose_bf16<<<gDD, blk, 0, stream>>>(Wg, wg16, DMODEL, DMODEL);
    cvt_transpose_bf16<<<gDD, blk, 0, stream>>>(Wo, wo16, DMODEL, DMODEL);
    dim3 gFC(DFF_ / 32, DMODEL / 32);         // Wfc: K=DMODEL, N=DFF
    cvt_transpose_bf16<<<gFC, blk, 0, stream>>>(Wfc, wfc16, DMODEL, DFF_);
    dim3 gPR(DMODEL / 32, DFF_ / 32);         // Wproj: K=DFF, N=DMODEL
    cvt_transpose_bf16<<<gPR, blk, 0, stream>>>(Wpr, wpr16, DFF_, DMODEL);
  }

  // 2) m0 = rms_norm(x) -> bf16
  rmsnorm_to_bf16<<<T_TOK, 256, 0, stream>>>(x, m0);

  // 3) q/k/v/g projections (WMMA bf16)
  dim3 blk(256);
  dim3 gNd(DMODEL / 128, T_TOK / 128);
  gemm_bf16_wmma<<<gNd, blk, 0, stream>>>(m0, wq16, qb, nullptr, nullptr, T_TOK, DMODEL, DMODEL, 0);
  gemm_bf16_wmma<<<gNd, blk, 0, stream>>>(m0, wk16, kb, nullptr, nullptr, T_TOK, DMODEL, DMODEL, 0);
  gemm_bf16_wmma<<<gNd, blk, 0, stream>>>(m0, wv16, vb, nullptr, nullptr, T_TOK, DMODEL, DMODEL, 0);
  gemm_bf16_wmma<<<gNd, blk, 0, stream>>>(m0, wg16, gb, nullptr, nullptr, T_TOK, DMODEL, DMODEL, 0);

  // 4) beta / alpha
  proj_beta_alpha<<<T_TOK, 256, 0, stream>>>(m0, Wb, Wa, A_log, dtb, beta, alph);

  // 5) activations: silu+l2norm for q,k; silu for v
  act_silu_l2<<<T_TOK * NHEAD, 128, 0, stream>>>(qb, 1);
  act_silu_l2<<<T_TOK * NHEAD, 128, 0, stream>>>(kb, 1);
  act_silu_l2<<<T_TOK * NHEAD, 128, 0, stream>>>(vb, 0);

  // 6) sequential gated delta-rule scan
  delta_scan<<<BATCH * NHEAD, 256, 0, stream>>>(qb, kb, vb, alph, beta, ob);

  // 7) gated RMSNorm -> bf16
  gated_rmsnorm<<<T_TOK * NHEAD, 128, 0, stream>>>(ob, gb, onw, og16);

  // 8) x1 = x + og @ Wo
  gemm_bf16_wmma<<<gNd, blk, 0, stream>>>(og16, wo16, x1, nullptr, x, T_TOK, DMODEL, DMODEL, 0);

  // 9) m1 = rms_norm(x1) -> bf16
  rmsnorm_to_bf16<<<T_TOK, 256, 0, stream>>>(x1, m1);

  // 10) ff = relu(m1 @ Wfc)^2 -> bf16 (fused epilogue)
  dim3 gFF(DFF_ / 128, T_TOK / 128);
  gemm_bf16_wmma<<<gFF, blk, 0, stream>>>(m1, wfc16, nullptr, ff16, nullptr, T_TOK, DFF_, DMODEL, 1);

  // 11) out = x1 + ff @ Wproj
  gemm_bf16_wmma<<<gNd, blk, 0, stream>>>(ff16, wpr16, out, nullptr, x1, T_TOK, DMODEL, DFF_, 0);
}
